// TemporalAttentionDecoder_44753559224629
// MI455X (gfx1250) — compile-verified, matrix operand order checked by reading, not run on hardware
//
#include <hip/hip_runtime.h>
#include <hip/hip_bf16.h>
#include <math.h>

// Problem sizes (fixed by the reference)
#define BSZ 512
#define TSZ 128
#define MSZ 512
#define PSZ 512

typedef __bf16 bf16_t;
typedef bf16_t v16bf __attribute__((ext_vector_type(16)));
typedef bf16_t v8bf  __attribute__((ext_vector_type(8)));
typedef bf16_t v4bf  __attribute__((ext_vector_type(4)));
typedef bf16_t v2bf  __attribute__((ext_vector_type(2)));
typedef float  v8f   __attribute__((ext_vector_type(8)));

// ---------------------------------------------------------------------------
// WMMA A-fragment loader (16x32 bf16 tile from row-major source, ld in elems):
//   lanes 0-15  : row = row0+lane,    elems 0..7 = K k0..k0+7,  8..15 = K k0+16..k0+23
//   lanes 16-31 : row = row0+lane-16, elems 0..7 = K k0+8..+15, 8..15 = K k0+24..+31
// The B-fragment (KxN, col-per-lane) of W^T equals the A-fragment of W, so all
// X @ W^T GEMMs load W tiles with this same loader (no transpose anywhere).
// ---------------------------------------------------------------------------
__device__ __forceinline__ v16bf frag_bf16(const bf16_t* p, int ld, int row0,
                                           int k0, int lane) {
  const bf16_t* q = p + (size_t)(row0 + (lane & 15)) * ld + k0 + ((lane >> 4) << 3);
  v8bf lo = *(const v8bf*)q;         // 16B vector loads (offsets 8-elem aligned)
  v8bf hi = *(const v8bf*)(q + 16);
  v16bf f;
#pragma unroll
  for (int i = 0; i < 8; ++i) { f[i] = lo[i]; f[i + 8] = hi[i]; }
  return f;
}

// ---------------------------------------------------------------------------
// Per-wave WMMA GEMM: C[Mt x N] = A[Mt x K] @ W[N x K]^T (+bias), C f32 or bf16.
// Each wave produces a 16x64 C tile (A frag reused across 4 WMMAs).
// Block = 256 threads = 8 waves -> 128 rows x 64 cols per block.
// grid = (Mt/128, N/64)
// ---------------------------------------------------------------------------
template <typename CT>
__global__ void gemm_wmma(const bf16_t* __restrict__ A, int lda,
                          const bf16_t* __restrict__ W,
                          const float* __restrict__ bias,
                          CT* __restrict__ C, int ldc, int K) {
  const int lane = threadIdx.x & 31;
  const int wave = threadIdx.x >> 5;
  const int row0 = (blockIdx.x * 8 + wave) * 16;
  const int col0 = blockIdx.y * 64;
  v8f acc[4] = {};
  for (int k0 = 0; k0 < K; k0 += 32) {
    v16bf a = frag_bf16(A, lda, row0, k0, lane);
#pragma unroll
    for (int j = 0; j < 4; ++j) {
      v16bf b = frag_bf16(W, K, col0 + 16 * j, k0, lane);
      acc[j] = __builtin_amdgcn_wmma_f32_16x16x32_bf16(
          false, a, false, b, (short)0, acc[j], false, false);
    }
  }
  const int nb = col0 + (lane & 15);
  const int rb = row0 + ((lane >> 4) << 3);  // lanes>=16 hold rows 8..15 of tile
#pragma unroll
  for (int j = 0; j < 4; ++j) {
    const int n = nb + 16 * j;
    const float bv = bias ? bias[n] : 0.f;
#pragma unroll
    for (int i = 0; i < 8; ++i)
      C[(size_t)(rb + i) * ldc + n] = (CT)(acc[j][i] + bv);
  }
}

// ---------------------------------------------------------------------------
// Helpers
// ---------------------------------------------------------------------------
__global__ void cvt_f32_to_bf16_x4(const float* __restrict__ src,
                                   bf16_t* __restrict__ dst, int n) {
  const int i = (blockIdx.x * blockDim.x + threadIdx.x) * 4;
  if (i < n) {
    const float4 v = *(const float4*)(src + i);
    v4bf o;
    o[0] = (bf16_t)v.x; o[1] = (bf16_t)v.y; o[2] = (bf16_t)v.z; o[3] = (bf16_t)v.w;
    *(v4bf*)(dst + i) = o;
  }
}

__device__ __forceinline__ float wave_sum(float v) {
#pragma unroll
  for (int o = 16; o > 0; o >>= 1) v += __shfl_xor(v, o, 32);
  return v;
}

__device__ __forceinline__ float sigmoidf(float x) { return 1.f / (1.f + expf(-x)); }

// l[b,t] = sum_m v_d[m] * tanh(x1[b,m] + Uy[b,t,m])  -- the hot kernel.
// grid = (T, B), block = 256; each thread handles 2 contiguous m (dword loads).
// Streams Uy (bf16, L2-resident); x1 row (2KB) stays in cache.
__global__ void attn_logits(const float* __restrict__ x1,
                            const bf16_t* __restrict__ Uy,
                            const float* __restrict__ vd,
                            float* __restrict__ l) {
  const int t = blockIdx.x, b = blockIdx.y;
  const int m0 = threadIdx.x * 2;                       // 256 threads * 2 = M
  const bf16_t* uy = Uy + ((size_t)b * TSZ + t) * MSZ;
  const float* x = x1 + (size_t)b * MSZ;
  const v2bf u = *(const v2bf*)(uy + m0);
  float acc = vd[m0]     * tanhf(x[m0]     + (float)u[0])
            + vd[m0 + 1] * tanhf(x[m0 + 1] + (float)u[1]);
  acc = wave_sum(acc);
  __shared__ float red[8];
  if ((threadIdx.x & 31) == 0) red[threadIdx.x >> 5] = acc;
  __syncthreads();
  if (threadIdx.x == 0) {
    float tot = 0.f;
#pragma unroll
    for (int i = 0; i < 8; ++i) tot += red[i];
    l[(size_t)b * TSZ + t] = tot;
  }
}

// beta[b,:] = softmax(l[b,:]) over T=128.  grid = B, block = 128.
__global__ void softmax_T(const float* __restrict__ l, float* __restrict__ beta) {
  const int b = blockIdx.x, tid = threadIdx.x;
  __shared__ float buf[TSZ];
  float v = l[(size_t)b * TSZ + tid];
  buf[tid] = v; __syncthreads();
  for (int s = 64; s > 0; s >>= 1) {
    if (tid < s) buf[tid] = fmaxf(buf[tid], buf[tid + s]);
    __syncthreads();
  }
  const float mx = buf[0]; __syncthreads();
  const float e = expf(v - mx);
  buf[tid] = e; __syncthreads();
  for (int s = 64; s > 0; s >>= 1) {
    if (tid < s) buf[tid] += buf[tid + s];
    __syncthreads();
  }
  beta[(size_t)b * TSZ + tid] = e / buf[0];
}

// c[b,m] = sum_t beta[b,t]*enc[b,t,m]; fused ytil[b] = c.w_tilda + y_t*w_M + b.
// grid = B, block = 256; each thread owns 2 contiguous m (dword bf16 loads,
// fully coalesced across the block). enc streamed as bf16 -> 64 MiB/step.
__global__ void context_ytilda(const float* __restrict__ beta,
                               const bf16_t* __restrict__ encb,
                               const float* __restrict__ y,
                               const float* __restrict__ wt,   // [M+1]
                               const float* __restrict__ wtb,  // [1]
                               float* __restrict__ c,
                               float* __restrict__ ytil, int tstep) {
  const int b = blockIdx.x;
  const int m0 = threadIdx.x * 2;
  const float* bb = beta + (size_t)b * TSZ;
  const bf16_t* eb = encb + (size_t)b * TSZ * MSZ + m0;
  float a0 = 0.f, a1 = 0.f;
#pragma unroll 4
  for (int t = 0; t < TSZ; ++t) {
    const v2bf e = *(const v2bf*)(eb + (size_t)t * MSZ);
    const float w = bb[t];
    a0 += w * (float)e[0];
    a1 += w * (float)e[1];
  }
  c[(size_t)b * MSZ + m0]     = a0;
  c[(size_t)b * MSZ + m0 + 1] = a1;
  float part = wave_sum(a0 * wt[m0] + a1 * wt[m0 + 1]);
  __shared__ float red[8];
  if ((threadIdx.x & 31) == 0) red[threadIdx.x >> 5] = part;
  __syncthreads();
  if (threadIdx.x == 0) {
    float tot = 0.f;
#pragma unroll
    for (int i = 0; i < 8; ++i) tot += red[i];
    ytil[b] = tot + y[(size_t)b * TSZ + tstep] * wt[MSZ] + wtb[0];
  }
}

// LSTM pointwise update; also writes bf16 [d|s] concat for next step's GEMMs.
// grid = B*P/256, block = 256.
__global__ void lstm_update(const float* __restrict__ ghh,   // [B,4P]
                            const float* __restrict__ ytil,  // [B]
                            const float* __restrict__ Wih,   // [4P]
                            const float* __restrict__ bih,
                            const float* __restrict__ bhh,
                            float* __restrict__ s, float* __restrict__ d,
                            bf16_t* __restrict__ cat_ds) {
  const int idx = blockIdx.x * blockDim.x + threadIdx.x;
  const int b = idx / PSZ, p = idx % PSZ;
  const float yt = ytil[b];
  const float* g = ghh + (size_t)b * 4 * PSZ;
  const float gi = sigmoidf(g[p]           + yt * Wih[p]           + bih[p]           + bhh[p]);
  const float gf = sigmoidf(g[PSZ + p]     + yt * Wih[PSZ + p]     + bih[PSZ + p]     + bhh[PSZ + p]);
  const float gg = tanhf   (g[2 * PSZ + p] + yt * Wih[2 * PSZ + p] + bih[2 * PSZ + p] + bhh[2 * PSZ + p]);
  const float go = sigmoidf(g[3 * PSZ + p] + yt * Wih[3 * PSZ + p] + bih[3 * PSZ + p] + bhh[3 * PSZ + p]);
  const float sn = gf * s[idx] + gi * gg;
  const float dn = go * tanhf(sn);
  s[idx] = sn;
  d[idx] = dn;
  cat_ds[(size_t)b * 2 * PSZ + p]       = (bf16_t)dn;
  cat_ds[(size_t)b * 2 * PSZ + PSZ + p] = (bf16_t)sn;
}

// out[b] = v_y . (W_y @ [d_T|c_T] + b_y) + v_y_b.  grid = B, block = 256.
__global__ void final_head(const float* __restrict__ d, const float* __restrict__ c,
                           const float* __restrict__ Wy,  // [P, P+M]
                           const float* __restrict__ Wyb,
                           const float* __restrict__ vy, const float* __restrict__ vyb,
                           float* __restrict__ out) {
  const int b = blockIdx.x;
  const float* db = d + (size_t)b * PSZ;
  const float* cb = c + (size_t)b * MSZ;
  float part = 0.f;
  for (int p = threadIdx.x; p < PSZ; p += 256) {
    const float* wr = Wy + (size_t)p * (PSZ + MSZ);
    float h = Wyb[p];
#pragma unroll 4
    for (int k = 0; k < PSZ; ++k) h += db[k] * wr[k];
#pragma unroll 4
    for (int k = 0; k < MSZ; ++k) h += cb[k] * wr[PSZ + k];
    part += h * vy[p];
  }
  part = wave_sum(part);
  __shared__ float red[8];
  if ((threadIdx.x & 31) == 0) red[threadIdx.x >> 5] = part;
  __syncthreads();
  if (threadIdx.x == 0) {
    float tot = 0.f;
#pragma unroll
    for (int i = 0; i < 8; ++i) tot += red[i];
    out[b] = tot + vyb[0];
  }
}

// ---------------------------------------------------------------------------
extern "C" void kernel_launch(void* const* d_in, const int* in_sizes, int n_in,
                              void* d_out, int out_size, void* d_ws, size_t ws_size,
                              hipStream_t stream) {
  (void)in_sizes; (void)n_in; (void)out_size; (void)ws_size;
  const float* enc   = (const float*)d_in[0];   // [B,T,M]
  const float* y     = (const float*)d_in[1];   // [B,T,1]
  const float* Wd    = (const float*)d_in[2];   // [M,2P]
  const float* Wdb   = (const float*)d_in[3];   // [M]
  const float* Ud    = (const float*)d_in[4];   // [M,M]
  const float* vd    = (const float*)d_in[5];   // [1,M]
  const float* wt    = (const float*)d_in[6];   // [1,M+1]
  const float* wtb   = (const float*)d_in[7];   // [1]
  const float* Wy    = (const float*)d_in[8];   // [P,P+M]
  const float* Wyb   = (const float*)d_in[9];   // [P]
  const float* vy    = (const float*)d_in[10];  // [1,P]
  const float* vyb   = (const float*)d_in[11];  // [1]
  const float* Wih   = (const float*)d_in[12];  // [4P,1]
  const float* Whh   = (const float*)d_in[13];  // [4P,P]
  const float* bih   = (const float*)d_in[14];  // [4P]
  const float* bhh   = (const float*)d_in[15];  // [4P]
  float* out = (float*)d_out;                   // [B,1]

  // Bump allocator over workspace (256B aligned).
  char* base = (char*)d_ws;
  size_t off = 0;
  auto alloc = [&](size_t bytes) -> char* {
    char* p = base + off;
    off += (bytes + 255) & ~(size_t)255;
    return p;
  };
  bf16_t* Ub     = (bf16_t*)alloc((size_t)MSZ * MSZ * 2);
  bf16_t* Wdb16  = (bf16_t*)alloc((size_t)MSZ * 2 * PSZ * 2);
  bf16_t* Whhb16 = (bf16_t*)alloc((size_t)4 * PSZ * PSZ * 2);
  bf16_t* encb   = (bf16_t*)alloc((size_t)BSZ * TSZ * MSZ * 2);   // 64 MiB
  bf16_t* Uy     = (bf16_t*)alloc((size_t)BSZ * TSZ * MSZ * 2);   // 64 MiB
  bf16_t* cat_ds = (bf16_t*)alloc((size_t)BSZ * 2 * PSZ * 2);
  float*  dbuf   = (float*)alloc((size_t)BSZ * PSZ * 4);
  float*  sbuf   = (float*)alloc((size_t)BSZ * PSZ * 4);
  float*  x1     = (float*)alloc((size_t)BSZ * MSZ * 4);
  float*  lbuf   = (float*)alloc((size_t)BSZ * TSZ * 4);
  float*  beta   = (float*)alloc((size_t)BSZ * TSZ * 4);
  float*  cbuf   = (float*)alloc((size_t)BSZ * MSZ * 4);
  float*  ytil   = (float*)alloc((size_t)BSZ * 4);
  float*  ghh    = (float*)alloc((size_t)BSZ * 4 * PSZ * 4);

  // 1) One-time bf16 conversions (weights + enc).
  cvt_f32_to_bf16_x4<<<(MSZ * MSZ / 4 + 255) / 256, 256, 0, stream>>>(Ud, Ub, MSZ * MSZ);
  cvt_f32_to_bf16_x4<<<(MSZ * 2 * PSZ / 4 + 255) / 256, 256, 0, stream>>>(Wd, Wdb16, MSZ * 2 * PSZ);
  cvt_f32_to_bf16_x4<<<(4 * PSZ * PSZ / 4 + 255) / 256, 256, 0, stream>>>(Whh, Whhb16, 4 * PSZ * PSZ);
  {
    const int n = BSZ * TSZ * MSZ;
    cvt_f32_to_bf16_x4<<<(n / 4 + 255) / 256, 256, 0, stream>>>(enc, encb, n);
  }

  // 2) Uy = enc @ U_d^T  (65536 x 512 x 512, WMMA bf16, bf16 output).
  gemm_wmma<bf16_t><<<dim3(BSZ * TSZ / 128, MSZ / 64), 256, 0, stream>>>(
      encb, MSZ, Ub, nullptr, Uy, MSZ, MSZ);

  // 3) Zero recurrent state.
  hipMemsetAsync(cat_ds, 0, (size_t)BSZ * 2 * PSZ * 2, stream);
  hipMemsetAsync(dbuf, 0, (size_t)BSZ * PSZ * 4, stream);
  hipMemsetAsync(sbuf, 0, (size_t)BSZ * PSZ * 4, stream);

  // 4) Recurrence over T timesteps (per-step streamed set ~128 MiB, L2-resident).
  for (int t = 0; t < TSZ; ++t) {
    // x1 = [d,s] @ W_d^T + b_d   (512x1024x512, WMMA)
    gemm_wmma<float><<<dim3(BSZ / 128, MSZ / 64), 256, 0, stream>>>(
        cat_ds, 2 * PSZ, Wdb16, Wdb, x1, MSZ, 2 * PSZ);
    // gates_hh = d @ W_hh^T       (512x512x2048, WMMA)
    gemm_wmma<float><<<dim3(BSZ / 128, 4 * PSZ / 64), 256, 0, stream>>>(
        cat_ds, 2 * PSZ, Whhb16, nullptr, ghh, 4 * PSZ, PSZ);
    // attention logits + softmax + context/ytilda
    attn_logits<<<dim3(TSZ, BSZ), 256, 0, stream>>>(x1, Uy, vd, lbuf);
    softmax_T<<<BSZ, TSZ, 0, stream>>>(lbuf, beta);
    context_ytilda<<<BSZ, 256, 0, stream>>>(beta, encb, y, wt, wtb, cbuf, ytil, t);
    // LSTM cell update
    lstm_update<<<BSZ * PSZ / 256, 256, 0, stream>>>(
        ghh, ytil, Wih, bih, bhh, sbuf, dbuf, cat_ds);
  }

  // 5) Final head.
  final_head<<<BSZ, 256, 0, stream>>>(dbuf, cbuf, Wy, Wyb, vy, vyb, out);
}
